// ADLS_41575283425677
// MI455X (gfx1250) — matrix operand order repaired and checked by reading, not compiled, not run on hardware
//
#include <hip/hip_runtime.h>

typedef __bf16 bf16;
typedef __attribute__((ext_vector_type(16))) __bf16 v16bf;
typedef __attribute__((ext_vector_type(8)))  float  v8f;

// ---------- sizes ----------
#define BSZ  16384
#define DD0  2048
#define DD1  2048
#define DD2  1024
#define DD3  512
#define NEXP 4
#define RNK  8
#define ER   32          // NEXP*RNK
#define MDOM 8
#define HDIM 64
#define NLAY 3

// GEMM tiling
#define BM   128
#define BN   128
#define BK   32
#define LDSK 40          // padded LDS row length (elements) -> conflict-free ds_load_b128

// ---------- WMMA helpers ----------
__device__ __forceinline__ v8f wmma_bf16(v16bf a, v16bf b, v8f c) {
    // (neg_a, A, neg_b, B, c_mod, C, reuse_a, reuse_b)
    return __builtin_amdgcn_wmma_f32_16x16x32_bf16(false, a, false, b, (short)0, c, false, false);
}

// A-operand fragment: lane holds row (lane&15); K elements {koff..koff+7} U {koff+16..koff+23}
// caller passes p already offset by row*ld + k0 + koff; we fetch p[0..7] and p[16..23].
__device__ __forceinline__ v16bf load_a(const bf16* p) {
    union { uint4 u[2]; v16bf v; } t;
    t.u[0] = *reinterpret_cast<const uint4*>(p);
    t.u[1] = *reinterpret_cast<const uint4*>(p + 16);
    return t.v;
}

// B-operand fragment: lane holds column (lane&15); K elements {koff..koff+7} U {koff+16..koff+23}, koff=(lane>=16)*8.
__device__ __forceinline__ v16bf load_b(const bf16* p) {
    union { uint4 u[2]; v16bf v; } t;
    t.u[0] = *reinterpret_cast<const uint4*>(p);
    t.u[1] = *reinterpret_cast<const uint4*>(p + 16);
    return t.v;
}

// ---------- CDNA5 async copy global -> LDS (ASYNCcnt-tracked, no VGPR round trip) ----------
// Generic pointers to LDS carry the workgroup-relative LDS byte offset in their low 32 bits
// (flat aperture mapping), which is exactly what VDST wants.
__device__ __forceinline__ void async_b128(const bf16* g, void* lds_ptr) {
    uint32_t l = (uint32_t)(uintptr_t)lds_ptr;
    asm volatile("global_load_async_to_lds_b128 %0, %1, off"
                 :: "v"(l), "v"(g) : "memory");
}
__device__ __forceinline__ void wait_async_le4() {
    asm volatile("s_wait_asynccnt 0x4" ::: "memory");
}
__device__ __forceinline__ void wait_async_0() {
    asm volatile("s_wait_asynccnt 0x0" ::: "memory");
}

// ---------- conversion kernels ----------
__global__ void __launch_bounds__(256)
cvt_f32_bf16(const float* __restrict__ src, bf16* __restrict__ dst, size_t n) {
    size_t i = (size_t)blockIdx.x * blockDim.x + threadIdx.x;
    size_t stride = (size_t)gridDim.x * blockDim.x;
    for (; i < n; i += stride) dst[i] = (bf16)src[i];
}

// Bsrc [E, N, R] f32 -> dst [N, ER] bf16 with dst[n, e*R+r] = Bsrc[e, n, r]
__global__ void __launch_bounds__(256)
build_bflat(const float* __restrict__ Bsrc, bf16* __restrict__ dst, int N) {
    size_t total = (size_t)N * ER;
    size_t i = (size_t)blockIdx.x * blockDim.x + threadIdx.x;
    size_t stride = (size_t)gridDim.x * blockDim.x;
    for (; i < total; i += stride) {
        int n = (int)(i >> 5);
        int c = (int)(i & 31);
        int e = c >> 3, r = c & 7;
        dst[i] = (bf16)Bsrc[((size_t)e * N + n) * RNK + r];
    }
}

// ---------- routing (tiny, one block) ----------
__global__ void __launch_bounds__(256)
routing_kernel(const float* __restrict__ dom_emb, const float* __restrict__ layer_pos,
               const float* __restrict__ Wi1, const float* __restrict__ bi1,
               const float* __restrict__ Wi2, const float* __restrict__ bi2,
               const float* __restrict__ Wa1, const float* __restrict__ ba1,
               const float* __restrict__ Wa2, const float* __restrict__ ba2,
               const float* __restrict__ gate_logits, const float* __restrict__ R_benefit,
               float* __restrict__ zetaAgg,   // [MDOM*NLAY]
               float* __restrict__ alphaAgg)  // [MDOM*NLAY*NEXP]
{
    const int tid = threadIdx.x;
    __shared__ float hz[MDOM * NLAY][HDIM];
    __shared__ float ha[MDOM * NLAY][HDIM];
    __shared__ float zl[MDOM * NLAY];
    __shared__ float al[MDOM * NLAY][NEXP];
    __shared__ float zall[MDOM * NLAY];
    __shared__ float aall[MDOM * NLAY][NEXP];
    __shared__ float Rn[MDOM][MDOM];

    // phase 1: the two router MLP hidden layers (24 rows x 64)
    for (int idx = tid; idx < MDOM * NLAY * HDIM; idx += 256) {
        int rowi = idx >> 6, o = idx & 63;
        int m = rowi / NLAY, l = rowi % NLAY;
        float sz = bi1[o], sa = ba1[o];
        for (int i = 0; i < HDIM; ++i) {
            float d = dom_emb[m * HDIM + i];
            float p = layer_pos[l * HDIM + i];
            sz += d * Wi1[o * 2 * HDIM + i] + p * Wi1[o * 2 * HDIM + HDIM + i];
            sa += d * Wa1[o * 2 * HDIM + i] + p * Wa1[o * 2 * HDIM + HDIM + i];
        }
        hz[rowi][o] = fmaxf(sz, 0.f);
        ha[rowi][o] = fmaxf(sa, 0.f);
    }
    __syncthreads();

    // phase 2: logits
    if (tid < MDOM * NLAY) {
        float s = bi2[0];
        for (int i = 0; i < HDIM; ++i) s += hz[tid][i] * Wi2[i];
        zl[tid] = s;
    } else if (tid < MDOM * NLAY + MDOM * NLAY * NEXP) {
        int t = tid - MDOM * NLAY;
        int rowi = t >> 2, e = t & 3;
        float s = ba2[e];
        for (int i = 0; i < HDIM; ++i) s += ha[rowi][i] * Wa2[e * HDIM + i];
        al[rowi][e] = s;
    }
    __syncthreads();

    // phase 3a: zeta top-2-of-3 sparse softmax per domain
    if (tid < MDOM) {
        float v[NLAY];
        for (int l = 0; l < NLAY; ++l) v[l] = zl[tid * NLAY + l];
        int drop = 0;  // smallest value, ties -> largest index (matches lax.top_k)
        for (int j = 1; j < NLAY; ++j)
            if (v[j] < v[drop] || (v[j] == v[drop] && j > drop)) drop = j;
        float mx = -3.4e38f;
        for (int l = 0; l < NLAY; ++l) if (l != drop) mx = fmaxf(mx, v[l]);
        float ex[NLAY], sum = 0.f;
        for (int l = 0; l < NLAY; ++l) { ex[l] = (l == drop) ? 0.f : __expf(v[l] - mx); sum += ex[l]; }
        for (int l = 0; l < NLAY; ++l) zall[tid * NLAY + l] = ex[l] / sum;
    }
    // phase 3b: alpha top-2-of-4 sparse softmax per (domain, layer)
    if (tid < MDOM * NLAY) {
        float v[NEXP];
        for (int e = 0; e < NEXP; ++e) v[e] = al[tid][e];
        int i1 = 0;
        for (int j = 1; j < NEXP; ++j) if (v[j] > v[i1]) i1 = j;
        int i2 = (i1 == 0) ? 1 : 0;
        for (int j = 0; j < NEXP; ++j) if (j != i1 && v[j] > v[i2]) i2 = j;
        float mx = fmaxf(v[i1], v[i2]);
        float e1 = __expf(v[i1] - mx), e2 = __expf(v[i2] - mx);
        float s = e1 + e2;
        for (int e = 0; e < NEXP; ++e) aall[tid][e] = 0.f;
        aall[tid][i1] = e1 / s;
        aall[tid][i2] = e2 / s;
    }
    // phase 3c: gated benefit matrix, row-normalized
    if (tid < MDOM) {
        float r[MDOM], s = 0.f;
        for (int n = 0; n < MDOM; ++n) {
            float sp = log1pf(__expf(gate_logits[tid * MDOM + n]));
            r[n] = sp * R_benefit[tid * MDOM + n];
            s += r[n];
        }
        s = fmaxf(s, 1e-12f);
        for (int n = 0; n < MDOM; ++n) Rn[tid][n] = r[n] / s;
    }
    __syncthreads();

    // phase 4: aggregation
    if (tid < MDOM * NLAY) {
        int m = tid / NLAY, l = tid % NLAY;
        float s = 0.f;
        for (int n = 0; n < MDOM; ++n) s += Rn[m][n] * zall[n * NLAY + l];
        zetaAgg[tid] = s;
    }
    if (tid < MDOM * NLAY * NEXP) {
        int m = tid / (NLAY * NEXP), rem = tid % (NLAY * NEXP);
        int l = rem >> 2, e = rem & 3;
        float s = 0.f;
        for (int n = 0; n < MDOM; ++n) s += Rn[m][n] * aall[n * NLAY + l][e];
        alphaAgg[tid] = s;
    }
}

// ---------- per-row weighted LoRA activation: tw[b, er] = zeta_b * alpha_{b,e} * (h_b . A_{er}) ----------
// WMMA GEMM, N = 32. Block = 8 waves, each wave 16 rows; block covers 128 rows.
__global__ void __launch_bounds__(256)
prep_tw_kernel(const bf16* __restrict__ h, const bf16* __restrict__ Aflat, int K,
               const int* __restrict__ dom,
               const float* __restrict__ zetaAgg, const float* __restrict__ alphaAgg,
               int layer, bf16* __restrict__ tw)
{
    const int lane = threadIdx.x & 31;
    const int wave = threadIdx.x >> 5;
    const int m0   = blockIdx.x * 128 + wave * 16;
    const int row  = lane & 15;
    const int koffA = (lane >> 4) * 8;
    const int koffB = (lane >> 4) * 8;

    v8f acc[2];
    v8f zero = {0.f, 0.f, 0.f, 0.f, 0.f, 0.f, 0.f, 0.f};
    acc[0] = zero; acc[1] = zero;

    const bf16* hp  = h + (size_t)(m0 + row) * K + koffA;
    const bf16* b0p = Aflat + (size_t)row * K + koffB;          // er cols 0..15
    const bf16* b1p = Aflat + (size_t)(16 + row) * K + koffB;   // er cols 16..31

    for (int k0 = 0; k0 < K; k0 += 32) {
        v16bf af = load_a(hp + k0);
        v16bf b0 = load_b(b0p + k0);
        v16bf b1 = load_b(b1p + k0);
        acc[0] = wmma_bf16(af, b0, acc[0]);
        acc[1] = wmma_bf16(af, b1, acc[1]);
    }

    const int mb = m0 + (lane >> 4) * 8;
#pragma unroll
    for (int j = 0; j < 2; ++j) {
        int n = j * 16 + row;     // er index 0..31
        int e = n >> 3;           // RNK == 8
#pragma unroll
        for (int v = 0; v < 8; ++v) {
            int m = mb + v;
            int d = dom[m];
            int base = d * NLAY + layer;
            float scale = zetaAgg[base] * alphaAgg[base * NEXP + e];
            tw[(size_t)m * ER + n] = (bf16)(acc[j][v] * scale);
        }
    }
}

// ---------- fused GEMM + LoRA tail + bias + ReLU, LDS-staged via async copies ----------
// out[m,n] = relu( A[m,:K].Wrow[n,:K] + tw[m,:32].Bflat[n,:32] + bias[n] )
// Block tile 128x128, 8 waves (4 along M x 2 along N), wave tile 32x64.
// Tiles double-buffered in LDS, filled with global_load_async_to_lds_b128 (ASYNCcnt).
__global__ void __launch_bounds__(256)
gemm_lora_relu(const bf16* __restrict__ A, const bf16* __restrict__ W,
               const float* __restrict__ bias,
               const bf16* __restrict__ tw, const bf16* __restrict__ Bflat,
               int K, int N,
               bf16* __restrict__ outBF, float* __restrict__ outF)
{
    __shared__ __align__(16) bf16 sA[2][BM * LDSK];
    __shared__ __align__(16) bf16 sB[2][BN * LDSK];

    const int tid  = threadIdx.x;
    const int lane = tid & 31;
    const int wave = tid >> 5;
    const int wm = wave >> 1;      // 0..3
    const int wn = wave & 1;       // 0..1
    const int blkM = blockIdx.y * BM;
    const int blkN = blockIdx.x * BN;
    const int row   = lane & 15;
    const int koffA = (lane >> 4) * 8;
    const int koffB = (lane >> 4) * 8;

    v8f zero = {0.f, 0.f, 0.f, 0.f, 0.f, 0.f, 0.f, 0.f};
    v8f acc[2][4];
#pragma unroll
    for (int i = 0; i < 2; ++i)
#pragma unroll
        for (int j = 0; j < 4; ++j) acc[i][j] = zero;

    // one K-chunk = BM x BK tile of A and BN x BK tile of W:
    // 512 16-byte segments each; 256 threads x 2 issues; seg = q*256+tid
    // -> consecutive threads fetch consecutive 16B (coalesced). 4 async ops/thread/chunk.
    auto issue_chunk = [&](int k0, int buf) {
#pragma unroll
        for (int q = 0; q < 2; ++q) {
            int seg = q * 256 + tid;
            int r = seg >> 2;
            int ks = (seg & 3) * 8;
            async_b128(A + (size_t)(blkM + r) * K + k0 + ks, &sA[buf][r * LDSK + ks]);
        }
#pragma unroll
        for (int q = 0; q < 2; ++q) {
            int seg = q * 256 + tid;
            int r = seg >> 2;
            int ks = (seg & 3) * 8;
            async_b128(W + (size_t)(blkN + r) * K + k0 + ks, &sB[buf][r * LDSK + ks]);
        }
    };

    auto compute_chunk = [&](int buf) {
        const bf16* a0 = &sA[buf][(wm * 32      + row) * LDSK + koffA];
        const bf16* a1 = &sA[buf][(wm * 32 + 16 + row) * LDSK + koffA];
        const bf16* b0 = &sB[buf][(wn * 64      + row) * LDSK + koffB];
        const bf16* b1 = &sB[buf][(wn * 64 + 16 + row) * LDSK + koffB];
        const bf16* b2 = &sB[buf][(wn * 64 + 32 + row) * LDSK + koffB];
        const bf16* b3 = &sB[buf][(wn * 64 + 48 + row) * LDSK + koffB];
        v16bf va0 = load_a(a0);
        v16bf va1 = load_a(a1);
        v16bf vb0 = load_b(b0);
        v16bf vb1 = load_b(b1);
        v16bf vb2 = load_b(b2);
        v16bf vb3 = load_b(b3);
        acc[0][0] = wmma_bf16(va0, vb0, acc[0][0]);
        acc[0][1] = wmma_bf16(va0, vb1, acc[0][1]);
        acc[0][2] = wmma_bf16(va0, vb2, acc[0][2]);
        acc[0][3] = wmma_bf16(va0, vb3, acc[0][3]);
        acc[1][0] = wmma_bf16(va1, vb0, acc[1][0]);
        acc[1][1] = wmma_bf16(va1, vb1, acc[1][1]);
        acc[1][2] = wmma_bf16(va1, vb2, acc[1][2]);
        acc[1][3] = wmma_bf16(va1, vb3, acc[1][3]);
    };

    const int nch = K / BK;
    issue_chunk(0, 0);
    for (int c = 0; c < nch; ++c) {
        if (c + 1 < nch) {
            issue_chunk((c + 1) * BK, (c + 1) & 1);  // prefetch next chunk (other buffer)
            wait_async_le4();                        // FIFO: chunk c's 4 ops have landed
        } else {
            wait_async_0();
        }
        __syncthreads();            // all waves' segments of chunk c visible
        compute_chunk(c & 1);
        __syncthreads();            // chunk c fully consumed before buffer reuse
    }

    // LoRA tail: one K=32 step against tw [BSZ,32] and Bflat [N,32] (tiny; direct global)
    {
        v16bf a0 = load_a(tw + (size_t)(blkM + wm * 32      + row) * ER + koffA);
        v16bf a1 = load_a(tw + (size_t)(blkM + wm * 32 + 16 + row) * ER + koffA);
        v16bf b0 = load_b(Bflat + (size_t)(blkN + wn * 64      + row) * ER + koffB);
        v16bf b1 = load_b(Bflat + (size_t)(blkN + wn * 64 + 16 + row) * ER + koffB);
        v16bf b2 = load_b(Bflat + (size_t)(blkN + wn * 64 + 32 + row) * ER + koffB);
        v16bf b3 = load_b(Bflat + (size_t)(blkN + wn * 64 + 48 + row) * ER + koffB);
        acc[0][0] = wmma_bf16(a0, b0, acc[0][0]);
        acc[0][1] = wmma_bf16(a0, b1, acc[0][1]);
        acc[0][2] = wmma_bf16(a0, b2, acc[0][2]);
        acc[0][3] = wmma_bf16(a0, b3, acc[0][3]);
        acc[1][0] = wmma_bf16(a1, b0, acc[1][0]);
        acc[1][1] = wmma_bf16(a1, b1, acc[1][1]);
        acc[1][2] = wmma_bf16(a1, b2, acc[1][2]);
        acc[1][3] = wmma_bf16(a1, b3, acc[1][3]);
    }

    // epilogue: C/D layout -> lane = col (lane&15), rows (lane>>4)*8 + v
    const int m0 = blkM + wm * 32;
    const int n0 = blkN + wn * 64;
#pragma unroll
    for (int j = 0; j < 4; ++j) {
        int n = n0 + j * 16 + row;
        float bv = bias[n];
#pragma unroll
        for (int i = 0; i < 2; ++i) {
            int mbase = m0 + i * 16 + (lane >> 4) * 8;
#pragma unroll
            for (int v = 0; v < 8; ++v) {
                float x = acc[i][j][v] + bv;
                x = x > 0.f ? x : 0.f;
                size_t idx = (size_t)(mbase + v) * N + n;
                if (outF)  outF[idx]  = x;
                if (outBF) outBF[idx] = (bf16)x;
            }
        }
    }
}

// ---------- host ----------
extern "C" void kernel_launch(void* const* d_in, const int* in_sizes, int n_in,
                              void* d_out, int out_size, void* d_ws, size_t ws_size,
                              hipStream_t stream) {
    (void)in_sizes; (void)n_in; (void)out_size; (void)ws_size;

    const float* x         = (const float*)d_in[0];
    const int*   dom       = (const int*)  d_in[1];
    const float* W1        = (const float*)d_in[2];
    const float* b1        = (const float*)d_in[3];
    const float* W2        = (const float*)d_in[4];
    const float* b2        = (const float*)d_in[5];
    const float* W3        = (const float*)d_in[6];
    const float* b3        = (const float*)d_in[7];
    const float* A1        = (const float*)d_in[8];
    const float* B1        = (const float*)d_in[9];
    const float* A2        = (const float*)d_in[10];
    const float* B2        = (const float*)d_in[11];
    const float* A3        = (const float*)d_in[12];
    const float* B3        = (const float*)d_in[13];
    const float* dom_emb   = (const float*)d_in[14];
    const float* layer_pos = (const float*)d_in[15];
    const float* Wi1       = (const float*)d_in[16];
    const float* bi1       = (const float*)d_in[17];
    const float* Wi2       = (const float*)d_in[18];
    const float* bi2       = (const float*)d_in[19];
    const float* Wa1       = (const float*)d_in[20];
    const float* ba1       = (const float*)d_in[21];
    const float* Wa2       = (const float*)d_in[22];
    const float* ba2       = (const float*)d_in[23];
    const float* gate      = (const float*)d_in[24];
    const float* Rben      = (const float*)d_in[25];
    float* out = (float*)d_out;

    char* ws = (char*)d_ws;
    size_t off = 0;
    auto carve = [&](size_t bytes) -> char* {
        char* p = ws + off;
        off += (bytes + 255) & ~(size_t)255;
        return p;
    };
    bf16* xbf  = (bf16*)carve((size_t)BSZ * DD0 * 2);
    bf16* h1bf = (bf16*)carve((size_t)BSZ * DD1 * 2);
    bf16* h2bf = (bf16*)carve((size_t)BSZ * DD2 * 2);
    bf16* w1bf = (bf16*)carve((size_t)DD1 * DD0 * 2);
    bf16* w2bf = (bf16*)carve((size_t)DD2 * DD1 * 2);
    bf16* w3bf = (bf16*)carve((size_t)DD3 * DD2 * 2);
    bf16* a1bf = (bf16*)carve((size_t)ER * DD0 * 2);
    bf16* a2bf = (bf16*)carve((size_t)ER * DD1 * 2);
    bf16* a3bf = (bf16*)carve((size_t)ER * DD2 * 2);
    bf16* bf1  = (bf16*)carve((size_t)DD1 * ER * 2);
    bf16* bf2  = (bf16*)carve((size_t)DD2 * ER * 2);
    bf16* bf3  = (bf16*)carve((size_t)DD3 * ER * 2);
    bf16* tw   = (bf16*)carve((size_t)BSZ * ER * 2);
    float* zetaAgg  = (float*)carve((size_t)MDOM * NLAY * 4);
    float* alphaAgg = (float*)carve((size_t)MDOM * NLAY * NEXP * 4);

    // precision staging: f32 -> bf16 (WMMA path)
    cvt_f32_bf16<<<4096, 256, 0, stream>>>(x,  xbf,  (size_t)BSZ * DD0);
    cvt_f32_bf16<<<1024, 256, 0, stream>>>(W1, w1bf, (size_t)DD1 * DD0);
    cvt_f32_bf16<<< 512, 256, 0, stream>>>(W2, w2bf, (size_t)DD2 * DD1);
    cvt_f32_bf16<<< 256, 256, 0, stream>>>(W3, w3bf, (size_t)DD3 * DD2);
    cvt_f32_bf16<<<  64, 256, 0, stream>>>(A1, a1bf, (size_t)ER * DD0);
    cvt_f32_bf16<<<  64, 256, 0, stream>>>(A2, a2bf, (size_t)ER * DD1);
    cvt_f32_bf16<<<  64, 256, 0, stream>>>(A3, a3bf, (size_t)ER * DD2);
    build_bflat<<<64, 256, 0, stream>>>(B1, bf1, DD1);
    build_bflat<<<64, 256, 0, stream>>>(B2, bf2, DD2);
    build_bflat<<<64, 256, 0, stream>>>(B3, bf3, DD3);

    routing_kernel<<<1, 256, 0, stream>>>(dom_emb, layer_pos, Wi1, bi1, Wi2, bi2,
                                          Wa1, ba1, Wa2, ba2, gate, Rben,
                                          zetaAgg, alphaAgg);

    // layer 1
    prep_tw_kernel<<<BSZ / 128, 256, 0, stream>>>(xbf, a1bf, DD0, dom, zetaAgg, alphaAgg, 0, tw);
    gemm_lora_relu<<<dim3(DD1 / BN, BSZ / BM), 256, 0, stream>>>(
        xbf, w1bf, b1, tw, bf1, DD0, DD1, h1bf, nullptr);
    // layer 2
    prep_tw_kernel<<<BSZ / 128, 256, 0, stream>>>(h1bf, a2bf, DD1, dom, zetaAgg, alphaAgg, 1, tw);
    gemm_lora_relu<<<dim3(DD2 / BN, BSZ / BM), 256, 0, stream>>>(
        h1bf, w2bf, b2, tw, bf2, DD1, DD2, h2bf, nullptr);
    // layer 3 (f32 output)
    prep_tw_kernel<<<BSZ / 128, 256, 0, stream>>>(h2bf, a3bf, DD2, dom, zetaAgg, alphaAgg, 2, tw);
    gemm_lora_relu<<<dim3(DD3 / BN, BSZ / BM), 256, 0, stream>>>(
        h2bf, w3bf, b3, tw, bf3, DD2, DD3, nullptr, out);
}
